// Layer3D_64149631533321
// MI455X (gfx1250) — compile-verified, matrix operand order checked by reading, not run on hardware
//
#include <hip/hip_runtime.h>
#include <math.h>

// ---------------------------------------------------------------- types
typedef __attribute__((ext_vector_type(16))) _Float16 v16h;
typedef __attribute__((ext_vector_type(8)))  float    v8f;

#define NEG_INF_V (-5000.0f)

__device__ __forceinline__ float gelu_exact(float x) {
    return 0.5f * x * (1.0f + erff(x * 0.70710678118654752440f));
}

// ---------------------------------------------------------------- WMMA grouped GEMM
// Y[img, g*Co+o, p] = sum_i X[imgX, g*Ci+i, p] * Wt[(g*Co+o)*Ci + i]
// rows r in [0,R): img = r/HW (X side remapped via imgStep/imgOff), p = r%HW.
// Each wave computes a 32(M)x16(N) tile: two A fragments share one B fragment.

template <bool KMASK, bool RMASK>
__device__ __forceinline__ void kblock_mm(
    const float* __restrict__ X, const float* __restrict__ Wt,
    long xB0, long xB1, long wB, int HW, int k0, int Ci,
    bool r0OK, bool r1OK, bool oOK, int hl,
    v8f& acc0, v8f& acc1)
{
    // ISA 16-bit A/B layout: lanes0-15 K{0-7,16-23}, lanes16-31 K{8-15,24-31}
    float a0f[16], a1f[16], bf[16];
#pragma unroll
    for (int e = 0; e < 16; ++e) {
        int k  = k0 + (e & 7) + ((e >> 3) << 4) + (hl << 3);
        int kc = KMASK ? ((k < Ci) ? k : (Ci - 1)) : k;
        a0f[e] = X[xB0 + (long)kc * HW];
        a1f[e] = X[xB1 + (long)kc * HW];
        bf[e]  = Wt[wB + kc];
    }
    v16h a0, a1, b;
#pragma unroll
    for (int e = 0; e < 16; ++e) {
        bool kok = KMASK ? ((k0 + (e & 7) + ((e >> 3) << 4) + (hl << 3)) < Ci) : true;
        bool m0 = RMASK ? (kok && r0OK) : kok;
        bool m1 = RMASK ? (kok && r1OK) : kok;
        bool mb = RMASK ? (kok && oOK)  : kok;
        a0[e] = (_Float16)(m0 ? a0f[e] : 0.0f);
        a1[e] = (_Float16)(m1 ? a1f[e] : 0.0f);
        b[e]  = (_Float16)(mb ? bf[e]  : 0.0f);
    }
    acc0 = __builtin_amdgcn_wmma_f32_16x16x32_f16(false, a0, false, b, (short)0, acc0, false, false);
    acc1 = __builtin_amdgcn_wmma_f32_16x16x32_f16(false, a1, false, b, (short)0, acc1, false, false);
}

__global__ void __launch_bounds__(256)
k_wmma_gemm(const float* __restrict__ X, const float* __restrict__ Wt,
            float* __restrict__ Y,
            int R, int HW, int G, int Ci, int Co,
            int CinT, int CoutT, int imgStep, int imgOff)
{
    const int Mt = (R + 31) >> 5;           // 32-row M blocks
    const int Nt = (Co + 15) >> 4;
    const int total = Mt * G * Nt;
    int wave = blockIdx.x * (blockDim.x >> 5) + (threadIdx.x >> 5);
    if (wave >= total) return;
    int mt = wave % Mt;
    int t2 = wave / Mt;
    int nt = t2 % Nt;
    int g  = t2 / Nt;

    int lane = threadIdx.x & 31;
    int hl   = lane >> 4;   // lane half selects K sub-range
    int l    = lane & 15;

    int r0 = mt * 32 + l;       // A rows (two sub-tiles)
    int r1 = r0 + 16;
    int o  = nt * 16 + l;       // B column (output channel in tile)
    bool r0OK = (r0 < R), r1OK = (r1 < R), oOK = (o < Co);

    int r0c = r0OK ? r0 : (R - 1);
    int r1c = r1OK ? r1 : (R - 1);
    int oc  = oOK  ? o  : (Co - 1);

    long xB0 = ((long)((r0c / HW) * imgStep + imgOff) * CinT + (long)g * Ci) * (long)HW + (r0c % HW);
    long xB1 = ((long)((r1c / HW) * imgStep + imgOff) * CinT + (long)g * Ci) * (long)HW + (r1c % HW);
    long wB  = (long)(g * Co + oc) * Ci;

    bool tileFull = (mt * 32 + 32 <= R) && (nt * 16 + 16 <= Co);  // wave-uniform
    int  kFull    = Ci & ~31;

    v8f acc0 = {}, acc1 = {};
    if (tileFull) {
        for (int k0 = 0; k0 < kFull; k0 += 32) {
            if (k0 + 32 < kFull)
                __builtin_prefetch((const void*)&Wt[wB + k0 + 32], 0, 1);
            kblock_mm<false, false>(X, Wt, xB0, xB1, wB, HW, k0, Ci,
                                    r0OK, r1OK, oOK, hl, acc0, acc1);
        }
    } else {
        for (int k0 = 0; k0 < kFull; k0 += 32) {
            kblock_mm<false, true>(X, Wt, xB0, xB1, wB, HW, k0, Ci,
                                   r0OK, r1OK, oOK, hl, acc0, acc1);
        }
    }
    if (kFull < Ci) {
        kblock_mm<true, true>(X, Wt, xB0, xB1, wB, HW, kFull, Ci,
                              r0OK, r1OK, oOK, hl, acc0, acc1);
    }

    // C/D layout: VGPR j -> (M = j + 8*hl, N = l)
#pragma unroll
    for (int j = 0; j < 8; ++j) {
        int rr = mt * 32 + j + (hl << 3);
        if (rr < R && oOK) {
            int img = rr / HW;
            int p   = rr % HW;
            Y[((long)img * CoutT + (g * Co + o)) * (long)HW + p] = acc0[j];
        }
        int rr1 = rr + 16;
        if (rr1 < R && oOK) {
            int img = rr1 / HW;
            int p   = rr1 % HW;
            Y[((long)img * CoutT + (g * Co + o)) * (long)HW + p] = acc1[j];
        }
    }
}

// ---------------------------------------------------------------- depthwise conv
// channels [c0,c1), weight Wd[(c-c0)*KH*KW + i*KW + j], zero padding
__global__ void k_dwconv(const float* __restrict__ X, const float* __restrict__ Wd,
                         float* __restrict__ Y, int Nimg, int C, int H, int W,
                         int c0, int c1, int KH, int KW, int padH, int padW)
{
    int HW = H * W;
    long total = (long)Nimg * (c1 - c0) * HW;
    long i = (long)blockIdx.x * blockDim.x + threadIdx.x;
    if (i >= total) return;
    int p  = i % HW;
    long t = i / HW;
    int ch = t % (c1 - c0);
    int n  = t / (c1 - c0);
    int c  = c0 + ch;
    int h  = p / W, w = p % W;
    float acc = 0.0f;
    for (int ii = 0; ii < KH; ++ii) {
        int hh = h + ii - padH;
        if (hh < 0 || hh >= H) continue;
        for (int jj = 0; jj < KW; ++jj) {
            int ww = w + jj - padW;
            if (ww < 0 || ww >= W) continue;
            acc += X[((long)n * C + c) * HW + hh * W + ww] * Wd[(ch * KH + ii) * KW + jj];
        }
    }
    Y[((long)n * C + c) * HW + p] = acc;
}

// ---------------------------------------------------------------- reductions
__global__ void k_mean_hw(const float* __restrict__ X, float* __restrict__ P, int HW)
{
    const float* xp = X + (long)blockIdx.x * HW;
    float s = 0.0f;
    for (int i = threadIdx.x; i < HW; i += blockDim.x) s += xp[i];
    __shared__ float sh[256];
    sh[threadIdx.x] = s; __syncthreads();
    for (int st = 128; st > 0; st >>= 1) {
        if (threadIdx.x < st) sh[threadIdx.x] += sh[threadIdx.x + st];
        __syncthreads();
    }
    if (threadIdx.x == 0) P[blockIdx.x] = sh[0] / (float)HW;
}

// vln stats per image: mean + 1/(std_{ddof=1}+eps)
__global__ void k_vln_stats(const float* __restrict__ X, float* __restrict__ stats, long per)
{
    long n = blockIdx.x;
    const float* xp = X + n * per;
    float s = 0.0f, ss = 0.0f;
    for (long i = threadIdx.x; i < per; i += blockDim.x) {
        float v = xp[i]; s += v; ss += v * v;
    }
    __shared__ float sh1[256], sh2[256];
    sh1[threadIdx.x] = s; sh2[threadIdx.x] = ss; __syncthreads();
    for (int st = 128; st > 0; st >>= 1) {
        if (threadIdx.x < st) {
            sh1[threadIdx.x] += sh1[threadIdx.x + st];
            sh2[threadIdx.x] += sh2[threadIdx.x + st];
        }
        __syncthreads();
    }
    if (threadIdx.x == 0) {
        float mu  = sh1[0] / (float)per;
        float var = (sh2[0] - mu * mu * (float)per) / (float)(per - 1);
        float sd  = sqrtf(fmaxf(var, 0.0f));
        stats[2 * n]     = mu;
        stats[2 * n + 1] = 1.0f / (sd + 1e-6f);
    }
}

__global__ void k_vln_apply(const float* __restrict__ X, const float* __restrict__ stats,
                            const float* __restrict__ gw, const float* __restrict__ bw,
                            float* __restrict__ Y, int C, int HW, long total,
                            int outStep, int outOff)
{
    long i = (long)blockIdx.x * blockDim.x + threadIdx.x;
    if (i >= total) return;
    int p  = i % HW;
    long t = i / HW;
    int c  = t % C;
    int n  = t / C;
    float mu = stats[2 * n], is = stats[2 * n + 1];
    float v = (X[i] - mu) * is * gw[c] + bw[c];
    Y[((long)(n * outStep + outOff) * C + c) * HW + p] = v;
}

// ---------------------------------------------------------------- channel attention
__global__ void k_ca_attn(const float* __restrict__ q, const float* __restrict__ k,
                          float* __restrict__ attnW, float scale)
{
    int t = threadIdx.x;
    if (t >= 128) return;                 // (b,h,n) = 2*8*8
    int b = t >> 6, r = t & 63, h = r >> 3, n = r & 7;
    float lg[8];
    for (int m = 0; m < 8; ++m) {
        if (m > n) { lg[m] = NEG_INF_V; continue; }
        float s = 0.0f;
        for (int d = 0; d < 48; ++d)
            s += q[(b * 8 + n) * 384 + h * 48 + d] * k[(b * 8 + m) * 384 + h * 48 + d];
        lg[m] = s * scale;
    }
    float mx = lg[0];
    for (int m = 1; m < 8; ++m) mx = fmaxf(mx, lg[m]);
    float w[8], sum = 0.0f;
    for (int m = 0; m < 8; ++m) { w[m] = expf(lg[m] - mx); sum += w[m]; }
    float inv = 1.0f / sum;
    for (int m = 0; m < 8; ++m) attnW[t * 8 + m] = w[m] * inv;
}

// bufA = x + einsum(attn, v)  (also the residual add of ln1 input)
__global__ void k_ca_out_add(const float* __restrict__ x, const float* __restrict__ V,
                             const float* __restrict__ attnW, float* __restrict__ Y, long total)
{
    long i = (long)blockIdx.x * blockDim.x + threadIdx.x;
    if (i >= total) return;
    const int C = 384, HW = 1024;
    int p  = i % HW;
    long t = i / HW;
    int c  = t % C;
    int n  = t / C;
    int b = n >> 3, nn = n & 7, h = c / 48;
    const float* aw = attnW + ((b * 8 + h) * 8 + nn) * 8;
    float acc = x[i];
    long vb = ((long)(b * 8) * C + c) * (long)HW + p;
    for (int m = 0; m < 8; ++m) acc += aw[m] * V[vb + (long)m * C * HW];
    Y[i] = acc;
}

// ---------------------------------------------------------------- full attention (n=1)
// Yq[b,c,p] = x[8b+7,c,p] + sum_m softmax(mask(q.k))_m * vh[8b+m,c,p]
__global__ void k_fa_out(const float* __restrict__ bufX, const float* __restrict__ qh,
                         const float* __restrict__ kh, const float* __restrict__ vh,
                         float* __restrict__ Yq, float scale)
{
    int t = blockIdx.x * blockDim.x + threadIdx.x;
    if (t >= 2 * 1024) return;
    const int C = 384, HW = 1024;
    int b = t / HW, p = t % HW;
    for (int h = 0; h < 8; ++h) {
        float lg[8];
        { // only m=0 unmasked (triu(ones(1,8),1) masks m>=1)
            float s = 0.0f;
            for (int d = 0; d < 48; ++d)
                s += qh[((long)b * C + h * 48 + d) * HW + p] *
                     kh[((long)(8 * b + 0) * C + h * 48 + d) * HW + p];
            lg[0] = s * scale;
        }
        for (int m = 1; m < 8; ++m) lg[m] = NEG_INF_V;
        float mx = lg[0];
        for (int m = 1; m < 8; ++m) mx = fmaxf(mx, lg[m]);
        float w[8], sum = 0.0f;
        for (int m = 0; m < 8; ++m) { w[m] = expf(lg[m] - mx); sum += w[m]; }
        float inv = 1.0f / sum;
        for (int d = 0; d < 48; ++d) {
            int c = h * 48 + d;
            float acc = bufX[((long)(8 * b + 7) * C + c) * HW + p];
            for (int m = 0; m < 8; ++m)
                acc += w[m] * inv * vh[((long)(8 * b + m) * C + c) * HW + p];
            Yq[((long)b * C + c) * HW + p] = acc;
        }
    }
}

// ---------------------------------------------------------------- elementwise
__global__ void k_add(float* __restrict__ Y, const float* __restrict__ A,
                      const float* __restrict__ B, long total)
{
    long i = (long)blockIdx.x * blockDim.x + threadIdx.x;
    if (i < total) Y[i] = A[i] + B[i];
}
__global__ void k_mul_gelu(float* __restrict__ Y, const float* __restrict__ A,
                           const float* __restrict__ B, long total)
{
    long i = (long)blockIdx.x * blockDim.x + threadIdx.x;
    if (i < total) Y[i] = A[i] * gelu_exact(B[i]);
}
__global__ void k_gelu_ip(float* __restrict__ Y, long total)
{
    long i = (long)blockIdx.x * blockDim.x + threadIdx.x;
    if (i < total) Y[i] = gelu_exact(Y[i]);
}
__global__ void k_relu_ip(float* __restrict__ Y, long total)
{
    long i = (long)blockIdx.x * blockDim.x + threadIdx.x;
    if (i < total) Y[i] = fmaxf(Y[i], 0.0f);
}
__global__ void k_sigmoid_ip(float* __restrict__ Y, long total)
{
    long i = (long)blockIdx.x * blockDim.x + threadIdx.x;
    if (i < total) Y[i] = 1.0f / (1.0f + expf(-Y[i]));
}
__global__ void k_se_scale(float* __restrict__ Y, const float* __restrict__ S,
                           int C, int HW, long total)
{
    long i = (long)blockIdx.x * blockDim.x + threadIdx.x;
    if (i >= total) return;
    long t = i / HW;
    int c = t % C;
    int n = t / C;
    Y[i] *= S[n * C + c];
}

// ---------------------------------------------------------------- host helpers
static inline unsigned blks(long n, int b) { return (unsigned)((n + b - 1) / b); }

static void gemm(const float* X, const float* Wt, float* Y, int R, int HW,
                 int G, int Ci, int Co, int CinT, int CoutT,
                 int imgStep, int imgOff, hipStream_t s)
{
    int Mt = (R + 31) / 32, Nt = (Co + 15) / 16;
    long waves = (long)Mt * G * Nt;
    k_wmma_gemm<<<blks(waves, 8), 256, 0, s>>>(X, Wt, Y, R, HW, G, Ci, Co,
                                               CinT, CoutT, imgStep, imgOff);
}

static void dwconv(const float* X, const float* Wd, float* Y, int Nimg, int C,
                   int c0, int c1, int KH, int KW, int padH, int padW, hipStream_t s)
{
    long total = (long)Nimg * (c1 - c0) * 1024;
    k_dwconv<<<blks(total, 256), 256, 0, s>>>(X, Wd, Y, Nimg, C, 32, 32,
                                              c0, c1, KH, KW, padH, padW);
}

// conv_block: returns pointer holding result (Yout if outw else T2)
static float* conv_block(const float* Xin, int din, int dout, int G, int K,
                         const float* pw1, const float* hwv, const float* wwv,
                         const float* pw2, const float* dwv, const float* outw,
                         float* T1, float* T2, float* Yout,
                         int Nimg, hipStream_t s)
{
    int R = Nimg * 1024, d1 = din / 2, pad = (K - 1) / 2;
    gemm(Xin, pw1, T1, R, 1024, G, din / G, din / G, din, din, 1, 0, s);
    dwconv(T1, hwv, T2, Nimg, din, 0, d1, K, 1, pad, 0, s);
    dwconv(T1, wwv, T2, Nimg, din, d1, din, 1, K, 0, pad, s);
    gemm(T2, pw2, T1, R, 1024, G, din / G, din / G, din, din, 1, 0, s);
    dwconv(T1, dwv, T2, Nimg, din, 0, din, 3, 3, 1, 1, s);
    if (outw) {
        gemm(T2, outw, Yout, R, 1024, 1, din, dout, din, dout, 1, 0, s);
        return Yout;
    }
    return T2;
}

// ---------------------------------------------------------------- input index map
enum {
    IN_X = 0, IN_CAQ, IN_CAK, IN_CAV, IN_FAQ, IN_FAK, IN_FAV,
    IN_LN1G, IN_LN1B, IN_LN2G, IN_LN2B, IN_LN3G, IN_LN3B,
    IN_WI1_PW1, IN_WI1_HW, IN_WI1_WW, IN_WI1_PW2, IN_WI1_DW, IN_WI1_OUT,
    IN_WI2_PW1, IN_WI2_HW, IN_WI2_WW, IN_WI2_PW2, IN_WI2_DW, IN_WI2_OUT,
    IN_WO_PW1, IN_WO_HW, IN_WO_WW, IN_WO_PW2, IN_WO_DW, IN_WO_OUT,
    IN_PC_PW1, IN_PC_HW, IN_PC_WW, IN_PC_PW2, IN_PC_DW,
    IN_PM_PW1, IN_PM_HW, IN_PM_WW, IN_PM_PW2, IN_PM_DW,
    IN_FC1, IN_FC2, IN_PLNG, IN_PLNB, IN_COUNT
};

extern "C" void kernel_launch(void* const* d_in, const int* in_sizes, int n_in,
                              void* d_out, int out_size, void* d_ws, size_t ws_size,
                              hipStream_t stream)
{
    const float* W[IN_COUNT];
    for (int i = 0; i < n_in && i < IN_COUNT; ++i) W[i] = (const float*)d_in[i];
    const float* x = W[IN_X];
    float* out = (float*)d_out;
    hipStream_t s = stream;

    const long E   = 2L * 8 * 384 * 1024;   // 6291456
    const long per = 384L * 1024;           // per-image elems for vln
    const float scale = 0.14433756729740643f; // 48^-0.5

    // ---- workspace carve
    char* base = (char*)d_ws;
    size_t off = 0;
    auto alloc = [&](long nfl) -> float* {
        float* r = (float*)(base + off);
        off += ((size_t)nfl * 4 + 255) & ~(size_t)255;
        return r;
    };
    float* bufA = alloc(E);       // residual sums / ffn out
    float* bufX = alloc(E);       // running x
    float* R1   = alloc(2 * E);   // V / kh ; ffn temp1
    float* R2   = alloc(2 * E);   // vh ; ffn temp2 ; post-conv result
    float* R3   = alloc(2 * E);   // ffn x1 ; mix temp/result
    float* R4   = alloc(2 * E);   // ffn x2
    float* qhB  = alloc(E / 8);
    float* bufQ = alloc(E / 8);
    float* pooled = alloc(6144);
    float* qm   = alloc(6144);
    float* km   = alloc(6144);
    float* attnW = alloc(1024);
    float* stats = alloc(64);
    float* sP   = alloc(6144);
    float* sY1  = alloc(512);
    float* sY2  = alloc(6144);
    (void)ws_size; (void)in_sizes; (void)out_size;

    // ================= channel attention =================
    k_mean_hw<<<16 * 384, 256, 0, s>>>(x, pooled, 1024);
    gemm(pooled, W[IN_CAQ], qm, 16, 1, 8, 48, 48, 384, 384, 1, 0, s);
    gemm(pooled, W[IN_CAK], km, 16, 1, 8, 48, 48, 384, 384, 1, 0, s);
    gemm(x, W[IN_CAV], R1, 16384, 1024, 8, 48, 48, 384, 384, 1, 0, s); // V
    k_ca_attn<<<1, 128, 0, s>>>(qm, km, attnW, scale);
    k_ca_out_add<<<blks(E, 256), 256, 0, s>>>(x, R1, attnW, bufA, E);
    k_vln_stats<<<16, 256, 0, s>>>(bufA, stats, per);
    k_vln_apply<<<blks(E, 256), 256, 0, s>>>(bufA, stats, W[IN_LN1G], W[IN_LN1B],
                                             bufX, 384, 1024, E, 1, 0);

    // ================= full attention (q = x[:, -1:]) =================
    gemm(bufX, W[IN_FAQ], qhB, 2048, 1024, 8, 48, 48, 384, 384, 8, 7, s);   // qh
    gemm(bufX, W[IN_FAK], R1, 16384, 1024, 8, 48, 48, 384, 384, 1, 0, s);   // kh
    gemm(bufX, W[IN_FAV], R2, 16384, 1024, 8, 48, 48, 384, 384, 1, 0, s);   // vh
    k_fa_out<<<8, 256, 0, s>>>(bufX, qhB, R1, R2, bufQ, scale);
    k_vln_stats<<<2, 256, 0, s>>>(bufQ, stats, per);
    k_vln_apply<<<blks(E / 8, 256), 256, 0, s>>>(bufQ, stats, W[IN_LN2G], W[IN_LN2B],
                                                 bufX, 384, 1024, E / 8, 8, 7); // write m=7 slice

    // ================= FFN =================
    conv_block(bufX, 384, 768, 8, 7, W[IN_WI1_PW1], W[IN_WI1_HW], W[IN_WI1_WW],
               W[IN_WI1_PW2], W[IN_WI1_DW], W[IN_WI1_OUT], R1, R2, R3, 16, s); // x1 -> R3
    conv_block(bufX, 384, 768, 8, 7, W[IN_WI2_PW1], W[IN_WI2_HW], W[IN_WI2_WW],
               W[IN_WI2_PW2], W[IN_WI2_DW], W[IN_WI2_OUT], R1, R2, R4, 16, s); // x2 -> R4
    k_mul_gelu<<<blks(2 * E, 256), 256, 0, s>>>(R3, R3, R4, 2 * E);
    conv_block(R3, 768, 384, 8, 7, W[IN_WO_PW1], W[IN_WO_HW], W[IN_WO_WW],
               W[IN_WO_PW2], W[IN_WO_DW], W[IN_WO_OUT], R1, R2, bufA, 16, s);  // ffn -> bufA
    k_add<<<blks(E, 256), 256, 0, s>>>(bufA, bufA, bufX, E);
    k_vln_stats<<<16, 256, 0, s>>>(bufA, stats, per);
    k_vln_apply<<<blks(E, 256), 256, 0, s>>>(bufA, stats, W[IN_LN3G], W[IN_LN3B],
                                             bufX, 384, 1024, E, 1, 0);

    // ================= layer2d =================
    k_add<<<blks(E, 256), 256, 0, s>>>(bufA, bufX, x, E);  // z = x + resid -> bufA
    float* cv = conv_block(bufA, 384, 384, 8, 7, W[IN_PC_PW1], W[IN_PC_HW], W[IN_PC_WW],
                           W[IN_PC_PW2], W[IN_PC_DW], nullptr, R1, R2, nullptr, 16, s); // -> R2
    k_gelu_ip<<<blks(E, 256), 256, 0, s>>>(cv, E);
    // squeeze-excite
    k_mean_hw<<<16 * 384, 256, 0, s>>>(cv, sP, 1024);
    gemm(sP, W[IN_FC1], sY1, 16, 1, 1, 384, 24, 384, 24, 1, 0, s);
    k_relu_ip<<<1, 256, 0, s>>>(sY1, 16 * 24);
    gemm(sY1, W[IN_FC2], sY2, 16, 1, 1, 24, 384, 24, 384, 1, 0, s);
    k_sigmoid_ip<<<24, 256, 0, s>>>(sY2, 16 * 384);
    k_se_scale<<<blks(E, 256), 256, 0, s>>>(cv, sY2, 384, 1024, E);
    // mix (g=1): temps R1,R3 (cv == R2 is the input)
    float* xv = conv_block(cv, 384, 384, 1, 7, W[IN_PM_PW1], W[IN_PM_HW], W[IN_PM_WW],
                           W[IN_PM_PW2], W[IN_PM_DW], nullptr, R1, R3, nullptr, 16, s); // -> R3
    k_add<<<blks(E, 256), 256, 0, s>>>(xv, xv, bufA, E);
    k_vln_stats<<<16, 256, 0, s>>>(xv, stats, per);
    k_vln_apply<<<blks(E, 256), 256, 0, s>>>(xv, stats, W[IN_PLNG], W[IN_PLNB],
                                             out, 384, 1024, E, 1, 0);
}
